// Attention_18837726560614
// MI455X (gfx1250) — matrix operand order repaired and checked by reading, not compiled, not run on hardware
//
#include <hip/hip_runtime.h>
#include <cstdint>
#include <cstddef>

// ---------------------------------------------------------------------------
// MHA forward for gfx1250 (MI455X): bf16 WMMA everywhere, flash-attention
// style streaming softmax. Attention stages K/V tiles into LDS with gfx1250
// async-LDS copies (ASYNCcnt) shared by all 8 waves of the workgroup.
// ---------------------------------------------------------------------------

#define BATCH 4
#define SEQ   2048
#define CH    1024
#define HEADS 16
#define DH    64
#define BN    (BATCH * SEQ)   // 8192 tokens
#define N3    (3 * CH)        // 3072 qkv columns

typedef __attribute__((ext_vector_type(16))) __bf16       v16bf;
typedef __attribute__((ext_vector_type(8)))  float        v8f;
typedef __attribute__((ext_vector_type(8)))  unsigned int v8u;

// f32 -> bf16 round-to-nearest-even (finite inputs)
static __device__ __forceinline__ unsigned short f2bf(float f) {
  unsigned u = __float_as_uint(f);
  return (unsigned short)((u + 0x7FFFu + ((u >> 16) & 1u)) >> 16);
}
static __device__ __forceinline__ unsigned pk2bf(float lo, float hi) {
  return (unsigned)f2bf(lo) | ((unsigned)f2bf(hi) << 16);
}

// B-frag (32x16 bf16): lane kHi*16+n holds 16 contiguous K halves -> 32B load
static __device__ __forceinline__ v16bf load_bfrag(const unsigned short* p) {
  return *reinterpret_cast<const v16bf*>(p);
}
// A-frag (16x32 bf16): lane h*16+m holds K = [16h,16h+8) and [16h+8,16h+16)
static __device__ __forceinline__ v16bf load_afrag(const unsigned short* p_lo,
                                                   const unsigned short* p_hi) {
  uint4 lo = *reinterpret_cast<const uint4*>(p_lo);
  uint4 hi = *reinterpret_cast<const uint4*>(p_hi);
  v8u u;
  u[0] = lo.x; u[1] = lo.y; u[2] = lo.z; u[3] = lo.w;
  u[4] = hi.x; u[5] = hi.y; u[6] = hi.z; u[7] = hi.w;
  return __builtin_bit_cast(v16bf, u);
}

#define WMMA_BF16(A, B, C) \
  __builtin_amdgcn_wmma_f32_16x16x32_bf16(false, (A), false, (B), (short)0, (C), false, false)

// ------------------------------- prep kernels ------------------------------

__global__ void k_cvt_x(const float* __restrict__ x, unsigned short* __restrict__ xb) {
  int gid = blockIdx.x * blockDim.x + threadIdx.x;       // one float4 each
  float4 f = reinterpret_cast<const float4*>(x)[gid];
  uint2 p; p.x = pk2bf(f.x, f.y); p.y = pk2bf(f.z, f.w);
  *reinterpret_cast<uint2*>(xb + (size_t)gid * 4) = p;
}

// w[K][Ncols] fp32 (row-major) -> wT[Ncols][K] bf16 (coalesced writes)
__global__ void k_cvt_transpose(const float* __restrict__ w, unsigned short* __restrict__ wT,
                                int K, int Ncols) {
  int idx = blockIdx.x * blockDim.x + threadIdx.x;
  int j  = idx / K;          // output row  = source column
  int kk = idx - j * K;      // output col  = source row
  wT[idx] = f2bf(w[(size_t)kk * Ncols + j]);
}

__global__ void k_maskbias(const unsigned char* __restrict__ mask, float* __restrict__ kbias) {
  int i = blockIdx.x * blockDim.x + threadIdx.x;
  kbias[i] = mask[i] ? -1.0e30f : 0.0f;
}

// ------------------------------- QKV GEMM ----------------------------------
// x_bf16[BN][CH] @ WqkvT^T + bqkv : wave computes a 64x64 output tile.
// Epilogue routes q/k -> [B,H,N,Dh], v -> transposed [B,H,Dh,N].

__global__ __launch_bounds__(128) void k_qkv_gemm(
    const unsigned short* __restrict__ xb, const unsigned short* __restrict__ wT,
    const float* __restrict__ bias,
    unsigned short* __restrict__ qout, unsigned short* __restrict__ kout,
    unsigned short* __restrict__ vtout) {
  const int ntiles = N3 / 64;                          // 48
  int wid  = blockIdx.x * 4 + (threadIdx.x >> 5);
  int lane = threadIdx.x & 31;
  int h = lane >> 4, ln = lane & 15;
  int m0 = (wid / ntiles) * 64;
  int n0 = (wid % ntiles) * 64;

  v8f acc[4][4];
  #pragma unroll
  for (int i = 0; i < 4; ++i)
    #pragma unroll
    for (int j = 0; j < 4; ++j) acc[i][j] = (v8f){};

  for (int k0 = 0; k0 < CH; k0 += 32) {
    v16bf a[4], b[4];
    #pragma unroll
    for (int i = 0; i < 4; ++i) {
      const unsigned short* row = xb + (size_t)(m0 + i * 16 + ln) * CH + k0;
      a[i] = load_afrag(row + 16 * h, row + 16 * h + 8);
    }
    #pragma unroll
    for (int j = 0; j < 4; ++j)
      b[j] = load_bfrag(wT + (size_t)(n0 + j * 16 + ln) * CH + k0 + 16 * h);
    #pragma unroll
    for (int i = 0; i < 4; ++i)
      #pragma unroll
      for (int j = 0; j < 4; ++j)
        acc[i][j] = WMMA_BF16(a[i], b[j], acc[i][j]);
  }

  #pragma unroll
  for (int j = 0; j < 4; ++j) {
    int col  = n0 + j * 16 + ln;
    float bv = bias[col];
    int sect = col >> 10;                // 0=q 1=k 2=v
    int cc   = col & (CH - 1);
    int head = cc >> 6;
    int d    = cc & (DH - 1);
    #pragma unroll
    for (int i = 0; i < 4; ++i) {
      int row = m0 + i * 16 + 8 * h;     // token for r=0
      int bb  = row >> 11;
      int tl  = row & (SEQ - 1);
      int bh  = bb * HEADS + head;
      if (sect == 0) {
        unsigned short* dst = qout + ((size_t)bh * SEQ + tl) * DH + d;
        #pragma unroll
        for (int r = 0; r < 8; ++r) dst[(size_t)r * DH] = f2bf((acc[i][j][r] + bv) * 0.125f);
      } else if (sect == 1) {
        unsigned short* dst = kout + ((size_t)bh * SEQ + tl) * DH + d;
        #pragma unroll
        for (int r = 0; r < 8; ++r) dst[(size_t)r * DH] = f2bf(acc[i][j][r] + bv);
      } else {                            // V transposed: 8 tokens contiguous
        uint4 pk;
        pk.x = pk2bf(acc[i][j][0] + bv, acc[i][j][1] + bv);
        pk.y = pk2bf(acc[i][j][2] + bv, acc[i][j][3] + bv);
        pk.z = pk2bf(acc[i][j][4] + bv, acc[i][j][5] + bv);
        pk.w = pk2bf(acc[i][j][6] + bv, acc[i][j][7] + bv);
        *reinterpret_cast<uint4*>(vtout + ((size_t)bh * DH + d) * SEQ + tl) = pk;
      }
    }
  }
}

// ------------------------------- attention ---------------------------------
// One block (8 waves) per (b,h, 128-query group); each wave owns 16 queries.
// K(32xDh) and V^T(Dhx32) tiles are staged into LDS once per block with
// gfx1250 async-LDS copies (double buffered on ASYNCcnt), then all waves run:
//   S^T(32x16) = K . Q^T        (4 WMMAs, A-frags from LDS)
//   online softmax over key rows (lane column = one query)
//   O^T(64x16) += V^T . P^T     (4 WMMAs, P^T built via shfl_xor 16)

#define KSTRIDE 72                      // 64 + 8 halves pad (bank-friendly)
#define VSTRIDE 40                      // 32 + 8 halves pad
#define KTILE (32 * KSTRIDE)            // halves
#define VTILE (64 * VSTRIDE)            // halves

__global__ __launch_bounds__(256) void k_attn(
    const unsigned short* __restrict__ q, const unsigned short* __restrict__ kmat,
    const unsigned short* __restrict__ vt, const float* __restrict__ kbias,
    unsigned short* __restrict__ att) {
  __shared__ unsigned short smem[2 * KTILE + 2 * VTILE];   // ~19 KB

  int tid  = threadIdx.x;
  int w    = tid >> 5;
  int lane = tid & 31;
  int h = lane >> 4, ln = lane & 15;
  int bh = blockIdx.x >> 4;                   // 64 (b,h) pairs
  int q0 = (blockIdx.x & 15) * 128 + w * 16;  // this wave's query tile
  int b = bh >> 4, head = bh & 15;

  const unsigned short* qbase = q    + (size_t)bh * SEQ * DH;
  const unsigned short* kbase = kmat + (size_t)bh * SEQ * DH;
  const unsigned short* vbase = vt   + (size_t)bh * DH * SEQ;
  const float* kbb = kbias + b * SEQ;

  // async staging assignment: K = 32 rows x 8 16B-chunks, V = 64 rows x 4
  unsigned lds0 = (unsigned)(size_t)smem;                     // LDS byte offset
  int krow_s = tid >> 3, kchk = tid & 7;
  int vrow_s = tid >> 2, vchk = tid & 3;
  const unsigned short* kgsrc = kbase + (size_t)krow_s * DH + kchk * 8;
  const unsigned short* vgsrc = vbase + (size_t)vrow_s * SEQ + vchk * 8;
  unsigned kldst = lds0 + (unsigned)(krow_s * KSTRIDE + kchk * 8) * 2;
  unsigned vldst = lds0 + (unsigned)(2 * KTILE + vrow_s * VSTRIDE + vchk * 8) * 2;

  auto stage = [&](int kb, int buf) {
    unsigned long long ka = (unsigned long long)(size_t)(kgsrc + (size_t)kb * DH);
    unsigned long long va = (unsigned long long)(size_t)(vgsrc + kb);
    asm volatile("global_load_async_to_lds_b128 %0, %1, off"
                 :: "v"(kldst + (unsigned)(buf * KTILE * 2)), "v"(ka) : "memory");
    asm volatile("global_load_async_to_lds_b128 %0, %1, off"
                 :: "v"(vldst + (unsigned)(buf * VTILE * 2)), "v"(va) : "memory");
  };

  // per-wave Q^T B-frags (dh 0..31 and 32..63)
  v16bf qb[2];
  #pragma unroll
  for (int c = 0; c < 2; ++c)
    qb[c] = load_bfrag(qbase + (size_t)(q0 + ln) * DH + c * 32 + 16 * h);

  v8f ot[4];
  #pragma unroll
  for (int dt = 0; dt < 4; ++dt) ot[dt] = (v8f){};
  float mrun = -3.0e38f, lrun = 0.0f;

  stage(0, 0);
  for (int kb = 0; kb < SEQ; kb += 32) {
    int cur = (kb >> 5) & 1;
    __syncthreads();                           // everyone done reading buf cur^1
    if (kb + 32 < SEQ) {
      stage(kb + 32, cur ^ 1);                 // prefetch next into other buffer
      asm volatile("s_wait_asynccnt 2" ::: "memory");   // current buffer landed
    } else {
      asm volatile("s_wait_asynccnt 0" ::: "memory");
    }
    __syncthreads();                           // current buffer visible to all

    const unsigned short* ks = smem + cur * KTILE;
    const unsigned short* vs = smem + 2 * KTILE + cur * VTILE;

    v8f s[2];
    #pragma unroll
    for (int t = 0; t < 2; ++t) {
      const unsigned short* krow = ks + (t * 16 + ln) * KSTRIDE;
      v16bf a0 = load_afrag(krow + 16 * h,      krow + 16 * h + 8);
      v16bf a1 = load_afrag(krow + 32 + 16 * h, krow + 40 + 16 * h);
      v8f z = (v8f){};
      z = WMMA_BF16(a0, qb[0], z);
      z = WMMA_BF16(a1, qb[1], z);
      s[t] = z;
    }
    // --- online softmax (column = query, rows = keys) ---
    float p0[8], p1[8];
    float vmax = -3.0e38f;
    #pragma unroll
    for (int r = 0; r < 8; ++r) {
      p0[r] = s[0][r] + kbb[kb + 8 * h + r];
      p1[r] = s[1][r] + kbb[kb + 16 + 8 * h + r];
      vmax = fmaxf(vmax, fmaxf(p0[r], p1[r]));
    }
    vmax = fmaxf(vmax, __shfl_xor(vmax, 16, 32));
    float mnew  = fmaxf(mrun, vmax);
    float alpha = __expf(mrun - mnew);
    float rsum  = 0.0f;
    #pragma unroll
    for (int r = 0; r < 8; ++r) {
      p0[r] = __expf(p0[r] - mnew);
      p1[r] = __expf(p1[r] - mnew);
      rsum += p0[r] + p1[r];
    }
    rsum += __shfl_xor(rsum, 16, 32);
    lrun = lrun * alpha + rsum;
    mrun = mnew;
    #pragma unroll
    for (int dt = 0; dt < 4; ++dt)
      #pragma unroll
      for (int r = 0; r < 8; ++r) ot[dt][r] *= alpha;

    // --- build P^T B-frag from the two S^T tiles via cross-half shuffles ---
    float e0[8], e1[8];
    #pragma unroll
    for (int r = 0; r < 8; ++r) {
      float o0 = __shfl_xor(p0[r], 16, 32);
      float o1 = __shfl_xor(p1[r], 16, 32);
      e0[r] = (h == 0) ? p0[r] : o1;   // frag elems 0..7  : keys 16h + 0..7
      e1[r] = (h == 0) ? o0 : p1[r];   // frag elems 8..15 : keys 16h + 8..15
    }
    v8u pu;
    #pragma unroll
    for (int rr = 0; rr < 4; ++rr) {
      pu[rr]     = pk2bf(e0[2 * rr], e0[2 * rr + 1]);
      pu[rr + 4] = pk2bf(e1[2 * rr], e1[2 * rr + 1]);
    }
    v16bf pb = __builtin_bit_cast(v16bf, pu);

    #pragma unroll
    for (int dt = 0; dt < 4; ++dt) {
      const unsigned short* vrow = vs + (dt * 16 + ln) * VSTRIDE;
      v16bf va = load_afrag(vrow + 16 * h, vrow + 16 * h + 8);
      ot[dt] = WMMA_BF16(va, pb, ot[dt]);
    }
  }

  float inv = 1.0f / lrun;
  int gt = b * SEQ + q0 + ln;                       // global token (= lane column)
  unsigned short* abase = att + (size_t)gt * CH + head * DH + 8 * h;
  #pragma unroll
  for (int dt = 0; dt < 4; ++dt) {
    uint4 pk;
    pk.x = pk2bf(ot[dt][0] * inv, ot[dt][1] * inv);
    pk.y = pk2bf(ot[dt][2] * inv, ot[dt][3] * inv);
    pk.z = pk2bf(ot[dt][4] * inv, ot[dt][5] * inv);
    pk.w = pk2bf(ot[dt][6] * inv, ot[dt][7] * inv);
    *reinterpret_cast<uint4*>(abase + dt * 16) = pk;
  }
}

// ------------------------------- output proj -------------------------------

__global__ __launch_bounds__(128) void k_proj_gemm(
    const unsigned short* __restrict__ att, const unsigned short* __restrict__ wT,
    const float* __restrict__ bias, float* __restrict__ out) {
  const int ntiles = CH / 64;                        // 16
  int wid  = blockIdx.x * 4 + (threadIdx.x >> 5);
  int lane = threadIdx.x & 31;
  int h = lane >> 4, ln = lane & 15;
  int m0 = (wid / ntiles) * 64;
  int n0 = (wid % ntiles) * 64;

  v8f acc[4][4];
  #pragma unroll
  for (int i = 0; i < 4; ++i)
    #pragma unroll
    for (int j = 0; j < 4; ++j) acc[i][j] = (v8f){};

  for (int k0 = 0; k0 < CH; k0 += 32) {
    v16bf a[4], b[4];
    #pragma unroll
    for (int i = 0; i < 4; ++i) {
      const unsigned short* row = att + (size_t)(m0 + i * 16 + ln) * CH + k0;
      a[i] = load_afrag(row + 16 * h, row + 16 * h + 8);
    }
    #pragma unroll
    for (int j = 0; j < 4; ++j)
      b[j] = load_bfrag(wT + (size_t)(n0 + j * 16 + ln) * CH + k0 + 16 * h);
    #pragma unroll
    for (int i = 0; i < 4; ++i)
      #pragma unroll
      for (int j = 0; j < 4; ++j)
        acc[i][j] = WMMA_BF16(a[i], b[j], acc[i][j]);
  }

  #pragma unroll
  for (int j = 0; j < 4; ++j) {
    int col  = n0 + j * 16 + ln;
    float bv = bias[col];
    #pragma unroll
    for (int i = 0; i < 4; ++i) {
      int row = m0 + i * 16 + 8 * h;
      #pragma unroll
      for (int r = 0; r < 8; ++r)
        out[(size_t)(row + r) * CH + col] = acc[i][j][r] + bv;
    }
  }
}

// ------------------------------- launch ------------------------------------

extern "C" void kernel_launch(void* const* d_in, const int* in_sizes, int n_in,
                              void* d_out, int out_size, void* d_ws, size_t ws_size,
                              hipStream_t stream) {
  (void)in_sizes; (void)n_in; (void)out_size; (void)ws_size;
  const float*         x     = (const float*)d_in[0];
  const unsigned char* mask  = (const unsigned char*)d_in[1];
  const float*         Wqkv  = (const float*)d_in[2];
  const float*         bqkv  = (const float*)d_in[3];
  const float*         Wproj = (const float*)d_in[4];
  const float*         bproj = (const float*)d_in[5];
  float*               out   = (float*)d_out;

  char* ws = (char*)d_ws;
  size_t off = 0;
  auto carve = [&](size_t bytes) -> char* {
    char* p = ws + off;
    off += (bytes + 255) & ~(size_t)255;
    return p;
  };
  unsigned short* xb     = (unsigned short*)carve((size_t)BN * CH * 2);      // 16 MB
  unsigned short* wqkvT  = (unsigned short*)carve((size_t)N3 * CH * 2);      //  6 MB
  unsigned short* wprojT = (unsigned short*)carve((size_t)CH * CH * 2);      //  2 MB
  float*          kbias  = (float*)carve((size_t)BN * 4);                    // 32 KB
  unsigned short* qbuf   = (unsigned short*)carve((size_t)BN * CH * 2);      // 16 MB
  unsigned short* kbuf   = (unsigned short*)carve((size_t)BN * CH * 2);      // 16 MB
  unsigned short* vtbuf  = (unsigned short*)carve((size_t)BN * CH * 2);      // 16 MB
  unsigned short* attbuf = (unsigned short*)carve((size_t)BN * CH * 2);      // 16 MB

  // prep
  k_cvt_x<<<(BN * CH / 4) / 256, 256, 0, stream>>>(x, xb);
  k_cvt_transpose<<<(N3 * CH) / 256, 256, 0, stream>>>(Wqkv, wqkvT, CH, N3);
  k_cvt_transpose<<<(CH * CH) / 256, 256, 0, stream>>>(Wproj, wprojT, CH, CH);
  k_maskbias<<<BN / 256, 256, 0, stream>>>(mask, kbias);

  // qkv projection (wave tiles 64x64 over 8192x3072)
  k_qkv_gemm<<<(BN / 64) * (N3 / 64) / 4, 128, 0, stream>>>(xb, wqkvT, bqkv,
                                                            qbuf, kbuf, vtbuf);
  // attention: one block per (b,h, 128 queries); K/V staged in LDS
  k_attn<<<BATCH * HEADS * (SEQ / 128), 256, 0, stream>>>(qbuf, kbuf, vtbuf,
                                                          kbias, attbuf);
  // output projection (wave tiles 64x64 over 8192x1024), fp32 out
  k_proj_gemm<<<(BN / 64) * (CH / 64) / 4, 128, 0, stream>>>(attbuf, wprojT, bproj, out);
}
